// AttentionLayer_Experts_16260746182880
// MI455X (gfx1250) — matrix operand order repaired and checked by reading, not compiled, not run on hardware
//
#include <hip/hip_runtime.h>
#include <hip/hip_bf16.h>

// ---------------------------------------------------------------------------
// MI455X (gfx1250, CDNA5). All matmuls via v_wmma_f32_16x16x32_bf16.
// GEMM kernels: block = 8 waves, block tile 256(M) x 64(N); per-wave 32x64
// (8 f32 accum tiles). B (weight) tile 64x32 staged cooperatively into LDS
// with double buffering via GLOBAL_LOAD_ASYNC_TO_LDS_B128 (ASYNCcnt), so the
// next chunk's copy is in flight while the current chunk's 8 WMMAs execute.
// A fragments are software-pipelined in registers.
// ---------------------------------------------------------------------------

typedef unsigned short u16;
typedef __attribute__((ext_vector_type(16))) __bf16 v16bf;
typedef __attribute__((ext_vector_type(8)))  __bf16 v8bf;
typedef __attribute__((ext_vector_type(8)))  float  v8f;
typedef __attribute__((ext_vector_type(8)))  unsigned short us8;
typedef __attribute__((ext_vector_type(4)))  int    v4i;

#define B_    2
#define S_    2048
#define D_    1024
#define H_    16
#define DA_   64
#define E_    4
#define DFF_  4096
#define SCALE_ (1.0f / (8.0f * 12.0f))   /* 1/(sqrt(64)*12) */
#define EPS_  1e-5f
#define NEG_  (-1.0e30f)

// round-to-nearest-even fp32 -> bf16
__device__ __forceinline__ u16 f2bf(float f) {
  unsigned u = __float_as_uint(f);
  u += 0x7FFFu + ((u >> 16) & 1u);
  return (u16)(u >> 16);
}

__device__ __forceinline__ v8f wmma32(v16bf a, v16bf b, v8f c) {
  return __builtin_amdgcn_wmma_f32_16x16x32_bf16(
      false, a, false, b, (short)0, c, false, false);
}

// A fragment (16x32 bf16, row-major source, base at tile row 0).
// lane m=l%16, half=l/16; K runs half*16+0..15 (contiguous).
__device__ __forceinline__ v16bf load_a(const u16* base, int ld, int kk, int lane) {
  int m  = lane & 15;
  int hf = lane >> 4;
  const __bf16* p = (const __bf16*)base + (size_t)m * ld + kk + hf * 16;
  union { v16bf v; v8bf h[2]; } u;
  u.h[0] = *(const v8bf*)p;
  u.h[1] = *(const v8bf*)(p + 8);
  return u.v;
}

// B fragment (32x16 bf16) from [N][K] layout (base at tile col 0).
// lane n=l%16 holds K = kk + half*16 + 0..15 (contiguous 32B).
__device__ __forceinline__ v16bf load_b(const u16* base, int ld, int kk, int lane) {
  int n  = lane & 15;
  int hf = lane >> 4;
  return *(const v16bf*)((const __bf16*)base + (size_t)n * ld + kk + hf * 16);
}

// 16-byte global -> LDS async copy (ASYNCcnt); sync fallback if unavailable.
__device__ __forceinline__ void cp_async16(const u16* g, u16* l) {
#if __has_builtin(__builtin_amdgcn_global_load_async_to_lds_b128)
  typedef __attribute__((address_space(1))) v4i* gp_t;
  typedef __attribute__((address_space(3))) v4i* lp_t;
  __builtin_amdgcn_global_load_async_to_lds_b128(
      (gp_t)const_cast<u16*>(g), (lp_t)l, 0, 0);
#else
  *(us8*)l = *(const us8*)g;
#endif
}

__device__ __forceinline__ void wait_async0() {
#if __has_builtin(__builtin_amdgcn_s_wait_asynccnt)
  __builtin_amdgcn_s_wait_asynccnt(0);
#else
  asm volatile("s_wait_asynccnt 0x0" ::: "memory");
#endif
}

// ---------------------------------------------------------------------------
// Shared GEMM main loop: C[256x64] += A[256xK] * B[Kx64].
// arow: A rows for this wave (2 row tiles: +0, +16). wcol: B in [N][K] bf16.
// B tile (64 cols x 32 K = 4KB) double-buffered in LDS, filled by async DMA.
// ---------------------------------------------------------------------------
__device__ __forceinline__ void gemm_mainloop_256x64(
    const u16* __restrict__ arow, int lda,
    const u16* __restrict__ wcol, int ldb,
    int nk32, int lane, int tid, v8f* acc /*[8]*/) {
  __shared__ __align__(32) u16 bbuf[2][64 * 32];

  const int fl = tid >> 2;            // 0..63 : B column
  const int dl = (tid & 3) * 8;       // 0,8,16,24 : K offset (8 bf16 = 16B)
  const u16* gsrc = wcol + (size_t)fl * ldb + dl;
  u16* l0 = &bbuf[0][fl * 32 + dl];
  u16* l1 = &bbuf[1][fl * 32 + dl];

  cp_async16(gsrc, l0);
  v16bf a0 = load_a(arow, lda, 0, lane);
  v16bf a1 = load_a(arow + (size_t)16 * lda, lda, 0, lane);
  wait_async0();
  __syncthreads();

  for (int it = 0; it < nk32; ++it) {
    // kick off next chunk's B-tile DMA into the other buffer
    if (it + 1 < nk32)
      cp_async16(gsrc + (size_t)(it + 1) * 32, (it & 1) ? l0 : l1);
    // prefetch next A fragments into separate registers
    v16bf a0n = a0, a1n = a1;
    if (it + 1 < nk32) {
      a0n = load_a(arow, lda, (it + 1) * 32, lane);
      a1n = load_a(arow + (size_t)16 * lda, lda, (it + 1) * 32, lane);
    }
    // all 4 B fragments first (one DS wait), then 8 back-to-back WMMAs
    const __bf16* bp = (const __bf16*)&bbuf[it & 1][0];
    v16bf bfr[4];
#pragma unroll
    for (int n = 0; n < 4; ++n)
      bfr[n] = *(const v16bf*)(bp + ((n * 16 + (lane & 15)) * 32 + (lane >> 4) * 16));
#pragma unroll
    for (int n = 0; n < 4; ++n) {
      acc[n]     = wmma32(a0, bfr[n], acc[n]);
      acc[4 + n] = wmma32(a1, bfr[n], acc[4 + n]);
    }
    a0 = a0n; a1 = a1n;
    wait_async0();
    __syncthreads();
  }
}

// ---------------------------------------------------------------------------
// helpers: fp32 -> bf16 convert, and fp32 [bt][R][C] -> bf16 [bt][C][R]
// ---------------------------------------------------------------------------
__global__ void f32_to_bf16_kernel(const float* __restrict__ in,
                                   u16* __restrict__ out, int n) {
  int i = blockIdx.x * 256 + threadIdx.x;
  if (i < n) out[i] = f2bf(in[i]);
}

__global__ void transpose_bf16_kernel(const float* __restrict__ in,
                                      u16* __restrict__ out,
                                      int R, int C, int nb) {
  int i = blockIdx.x * 256 + threadIdx.x;
  int total = nb * R * C;
  if (i >= total) return;
  int bt  = i / (R * C);
  int rem = i - bt * (R * C);
  int r = rem / C;
  int c = rem - r * C;
  out[(size_t)bt * R * C + (size_t)c * R + r] = f2bf(in[i]);
}

// ---------------------------------------------------------------------------
// QKV projection: per head GEMM [4096 x 1024] x [1024 x 64].
// grid = (4096/256, H, 3). q,k stored [B,H,S,64]; v transposed [B,H,64,S].
// ---------------------------------------------------------------------------
__global__ __launch_bounds__(256) void qkv_kernel(
    const u16* __restrict__ xb, const u16* __restrict__ wqt,
    const u16* __restrict__ wkt, const u16* __restrict__ wvt,
    const float* __restrict__ bq, const float* __restrict__ bk,
    const float* __restrict__ bv, u16* __restrict__ qb,
    u16* __restrict__ kb, u16* __restrict__ vtb) {
  int tid = threadIdx.x, lane = tid & 31, wave = tid >> 5;
  int h = blockIdx.y, which = blockIdx.z;
  int tok0 = blockIdx.x * 256 + wave * 32;
  const u16*  wt   = (which == 0) ? wqt : (which == 1) ? wkt : wvt;
  const float* bias = (which == 0) ? bq : (which == 1) ? bk : bv;
  const u16* wh = wt + (size_t)h * DA_ * D_;   // [DA][D]

  v8f zero = {};
  v8f acc[8];
#pragma unroll
  for (int n = 0; n < 8; ++n) acc[n] = zero;

  gemm_mainloop_256x64(xb + (size_t)tok0 * D_, D_, wh, D_, D_ / 32, lane, tid, acc);

  int hf = lane >> 4, nn = lane & 15;
#pragma unroll
  for (int rt = 0; rt < 2; ++rt) {
#pragma unroll
    for (int n = 0; n < 4; ++n) {
      int e = n * 16 + nn;
      float bia = bias[h * DA_ + e];
#pragma unroll
      for (int r = 0; r < 8; ++r) {
        int tok = tok0 + rt * 16 + hf * 8 + r;
        int b = tok >> 11, s = tok & (S_ - 1);
        float val = acc[rt * 4 + n][r] + bia;
        if (which == 2) {
          vtb[((size_t)(b * H_ + h) * DA_ + e) * S_ + s] = f2bf(val);
        } else {
          u16* dst = (which == 0) ? qb : kb;
          dst[((size_t)(b * H_ + h) * S_ + s) * DA_ + e] = f2bf(val);
        }
      }
    }
  }
}

// ---------------------------------------------------------------------------
// Flash-style causal attention. Block = (b,h,128 query rows); 8 waves each
// own 16 query rows. 32-wide key blocks; online softmax; P staged through a
// per-wave LDS tile. K/V fragments batched at loop top so V loads overlap
// score WMMAs and softmax VALU.
// ---------------------------------------------------------------------------
__global__ __launch_bounds__(256) void attn_kernel(
    const u16* __restrict__ qb, const u16* __restrict__ kb,
    const u16* __restrict__ vtb, float* __restrict__ attn_out) {
  __shared__ __align__(32) u16 pbuf[8][16 * 32];

  int lane = threadIdx.x & 31, wave = threadIdx.x >> 5;
  int bh = blockIdx.x >> 4, qblk = blockIdx.x & 15;
  int b = bh >> 4, h = bh & 15;
  int s0 = qblk * 128 + wave * 16;
  int hf = lane >> 4, nn = lane & 15;

  const u16* qh = qb  + (size_t)bh * S_ * DA_;
  const u16* kh = kb  + (size_t)bh * S_ * DA_;
  const u16* vh = vtb + (size_t)bh * DA_ * S_;   // [64][S]

  v16bf aq[2];
  aq[0] = load_a(qh + (size_t)s0 * DA_, DA_, 0, lane);
  aq[1] = load_a(qh + (size_t)s0 * DA_, DA_, 32, lane);

  v8f zero = {};
  v8f o[4];
#pragma unroll
  for (int k = 0; k < 4; ++k) o[k] = zero;
  float m8[8], l8[8];
#pragma unroll
  for (int r = 0; r < 8; ++r) { m8[r] = NEG_; l8[r] = 0.0f; }

  int tblocks = (s0 + 16 + 31) >> 5;   // causal upper bound
  for (int tb = 0; tb < tblocks; ++tb) {
    int t0 = tb * 32;
    // batch-issue all V and K fragment loads for this key block
    v16bf bvf[4];
#pragma unroll
    for (int k = 0; k < 4; ++k)
      bvf[k] = load_b(vh + (size_t)(k * 16) * S_, S_, t0, lane);
    v16bf bkf[4];
#pragma unroll
    for (int j = 0; j < 2; ++j)
#pragma unroll
      for (int cc = 0; cc < 2; ++cc)
        bkf[j * 2 + cc] =
            load_b(kh + (size_t)(t0 + j * 16) * DA_, DA_, cc * 32, lane);
    // scores: two 16x16 sub-tiles
    v8f sc[2];
#pragma unroll
    for (int j = 0; j < 2; ++j) {
      v8f c = zero;
#pragma unroll
      for (int cc = 0; cc < 2; ++cc) c = wmma32(aq[cc], bkf[j * 2 + cc], c);
      sc[j] = c;
    }
    // mask + scale + online softmax update (row = hf*8 + r, col = n)
#pragma unroll
    for (int r = 0; r < 8; ++r) {
      int srow = s0 + hf * 8 + r;
      float v0 = (t0 + nn      <= srow) ? sc[0][r] * SCALE_ : NEG_;
      float v1 = (t0 + 16 + nn <= srow) ? sc[1][r] * SCALE_ : NEG_;
      float mx = fmaxf(v0, v1);
#pragma unroll
      for (int msk = 1; msk < 16; msk <<= 1)
        mx = fmaxf(mx, __shfl_xor(mx, msk, 32));
      float mnew  = fmaxf(m8[r], mx);
      float scale = __expf(m8[r] - mnew);
      m8[r] = mnew;
      float p0 = __expf(v0 - mnew);
      float p1 = __expf(v1 - mnew);
      float rs = p0 + p1;
#pragma unroll
      for (int msk = 1; msk < 16; msk <<= 1)
        rs += __shfl_xor(rs, msk, 32);
      l8[r] = l8[r] * scale + rs;
#pragma unroll
      for (int k = 0; k < 4; ++k) o[k][r] *= scale;
      int row = hf * 8 + r;
      pbuf[wave][row * 32 + nn]      = f2bf(p0);
      pbuf[wave][row * 32 + 16 + nn] = f2bf(p1);
    }
    asm volatile("s_wait_dscnt 0x0" ::: "memory");
    __builtin_amdgcn_wave_barrier();
    // re-load P as an A fragment (16x32)
    union { v16bf v; v8bf h2[2]; } up;
    const __bf16* pp = (const __bf16*)&pbuf[wave][0];
    up.h2[0] = *(const v8bf*)(pp + nn * 32 + hf * 16);
    up.h2[1] = *(const v8bf*)(pp + nn * 32 + hf * 16 + 8);
#pragma unroll
    for (int k = 0; k < 4; ++k) o[k] = wmma32(up.v, bvf[k], o[k]);
    __builtin_amdgcn_wave_barrier();
  }

  // normalize and scatter into [B,S,D] (heads concatenated)
#pragma unroll
  for (int r = 0; r < 8; ++r) {
    float inv = 1.0f / l8[r];
    int s = s0 + hf * 8 + r;
#pragma unroll
    for (int k = 0; k < 4; ++k) {
      attn_out[((size_t)b * S_ + s) * D_ + h * DA_ + k * 16 + nn] = o[k][r] * inv;
    }
  }
}

// ---------------------------------------------------------------------------
// LayerNorm 1: x1 = LN(x + attn*scalar); writes fp32 + bf16 copies.
// ---------------------------------------------------------------------------
__global__ __launch_bounds__(256) void ln1_kernel(
    const float* __restrict__ x, const float* __restrict__ ao,
    const float* __restrict__ scalar, const float* __restrict__ g,
    const float* __restrict__ bta, float* __restrict__ x1f,
    u16* __restrict__ x1b) {
  __shared__ float red[256];
  int tok = blockIdx.x, tid = threadIdx.x;
  const float* xr = x  + (size_t)tok * D_;
  const float* ar = ao + (size_t)tok * D_;
  float sc = scalar[0];
  float v[4], sum = 0.0f, sq = 0.0f;
#pragma unroll
  for (int i = 0; i < 4; ++i) {
    int d = tid + i * 256;
    float t = xr[d] + ar[d] * sc;
    v[i] = t; sum += t; sq += t * t;
  }
  red[tid] = sum; __syncthreads();
  for (int off = 128; off; off >>= 1) {
    if (tid < off) red[tid] += red[tid + off];
    __syncthreads();
  }
  float mu = red[0] * (1.0f / D_);
  __syncthreads();
  red[tid] = sq; __syncthreads();
  for (int off = 128; off; off >>= 1) {
    if (tid < off) red[tid] += red[tid + off];
    __syncthreads();
  }
  float var  = red[0] * (1.0f / D_) - mu * mu;
  float rstd = rsqrtf(var + EPS_);
#pragma unroll
  for (int i = 0; i < 4; ++i) {
    int d = tid + i * 256;
    float y = (v[i] - mu) * rstd * g[d] + bta[d];
    x1f[(size_t)tok * D_ + d] = y;
    x1b[(size_t)tok * D_ + d] = f2bf(y);
  }
}

// ---------------------------------------------------------------------------
// Expert FFN GEMM 1: h = gelu(x1 @ W1 + b1). grid (DFF/64, 512/256, B*E).
// ---------------------------------------------------------------------------
__global__ __launch_bounds__(256) void expert1_kernel(
    const u16* __restrict__ x1b, const u16* __restrict__ w1t,
    const float* __restrict__ eb1, u16* __restrict__ hb) {
  int tid = threadIdx.x, lane = tid & 31, wave = tid >> 5;
  int chunk = blockIdx.z;                  // b*E + e; tokens chunk*512+i
  int e = chunk & 3;
  int tok0 = chunk * 512 + blockIdx.y * 256 + wave * 32;
  int col0 = blockIdx.x * 64;
  const u16* wcol = w1t + (size_t)e * DFF_ * D_ + (size_t)col0 * D_;  // [f][d]

  v8f zero = {};
  v8f acc[8];
#pragma unroll
  for (int n = 0; n < 8; ++n) acc[n] = zero;

  gemm_mainloop_256x64(x1b + (size_t)tok0 * D_, D_, wcol, D_, D_ / 32,
                       lane, tid, acc);

  int hf = lane >> 4, nn = lane & 15;
#pragma unroll
  for (int rt = 0; rt < 2; ++rt) {
#pragma unroll
    for (int n = 0; n < 4; ++n) {
      int f = col0 + n * 16 + nn;
      float bia = eb1[e * DFF_ + f];
#pragma unroll
      for (int r = 0; r < 8; ++r) {
        int tok = tok0 + rt * 16 + hf * 8 + r;
        float val = acc[rt * 4 + n][r] + bia;
        float gl  = 0.5f * val * (1.0f + erff(val * 0.70710678118654752f));
        hb[(size_t)tok * DFF_ + f] = f2bf(gl);
      }
    }
  }
}

// ---------------------------------------------------------------------------
// Expert FFN GEMM 2: y = h @ W2 + b2 (fp32 out). grid (D/64, 512/256, B*E).
// ---------------------------------------------------------------------------
__global__ __launch_bounds__(256) void expert2_kernel(
    const u16* __restrict__ hb, const u16* __restrict__ w2t,
    const float* __restrict__ eb2, float* __restrict__ yf) {
  int tid = threadIdx.x, lane = tid & 31, wave = tid >> 5;
  int chunk = blockIdx.z;
  int e = chunk & 3;
  int tok0 = chunk * 512 + blockIdx.y * 256 + wave * 32;
  int col0 = blockIdx.x * 64;
  const u16* wcol = w2t + (size_t)e * D_ * DFF_ + (size_t)col0 * DFF_;  // [d][f]

  v8f zero = {};
  v8f acc[8];
#pragma unroll
  for (int n = 0; n < 8; ++n) acc[n] = zero;

  gemm_mainloop_256x64(hb + (size_t)tok0 * DFF_, DFF_, wcol, DFF_, DFF_ / 32,
                       lane, tid, acc);

  int hf = lane >> 4, nn = lane & 15;
#pragma unroll
  for (int rt = 0; rt < 2; ++rt) {
#pragma unroll
    for (int n = 0; n < 4; ++n) {
      int d = col0 + n * 16 + nn;
      float bia = eb2[e * D_ + d];
#pragma unroll
      for (int r = 0; r < 8; ++r) {
        int tok = tok0 + rt * 16 + hf * 8 + r;
        yf[(size_t)tok * D_ + d] = acc[rt * 4 + n][r] + bia;
      }
    }
  }
}

// ---------------------------------------------------------------------------
// LayerNorm 2 (per-expert params): out = LN(x1 + y*e_scalar[e])
// ---------------------------------------------------------------------------
__global__ __launch_bounds__(256) void ln2_kernel(
    const float* __restrict__ x1f, const float* __restrict__ yf,
    const float* __restrict__ e_scalar, const float* __restrict__ eg,
    const float* __restrict__ ebt, float* __restrict__ out) {
  __shared__ float red[256];
  int tok = blockIdx.x, tid = threadIdx.x;
  int e = (tok & (S_ - 1)) >> 9;
  const float* xr = x1f + (size_t)tok * D_;
  const float* yr = yf  + (size_t)tok * D_;
  float sc = e_scalar[e];
  float v[4], sum = 0.0f, sq = 0.0f;
#pragma unroll
  for (int i = 0; i < 4; ++i) {
    int d = tid + i * 256;
    float t = xr[d] + yr[d] * sc;
    v[i] = t; sum += t; sq += t * t;
  }
  red[tid] = sum; __syncthreads();
  for (int off = 128; off; off >>= 1) {
    if (tid < off) red[tid] += red[tid + off];
    __syncthreads();
  }
  float mu = red[0] * (1.0f / D_);
  __syncthreads();
  red[tid] = sq; __syncthreads();
  for (int off = 128; off; off >>= 1) {
    if (tid < off) red[tid] += red[tid + off];
    __syncthreads();
  }
  float var  = red[0] * (1.0f / D_) - mu * mu;
  float rstd = rsqrtf(var + EPS_);
#pragma unroll
  for (int i = 0; i < 4; ++i) {
    int d = tid + i * 256;
    out[(size_t)tok * D_ + d] = (v[i] - mu) * rstd * eg[e * D_ + d] + ebt[e * D_ + d];
  }
}

// ---------------------------------------------------------------------------
extern "C" void kernel_launch(void* const* d_in, const int* in_sizes, int n_in,
                              void* d_out, int out_size, void* d_ws, size_t ws_size,
                              hipStream_t stream) {
  (void)in_sizes; (void)n_in; (void)out_size; (void)ws_size;
  const float* x        = (const float*)d_in[0];
  const float* Wq       = (const float*)d_in[1];
  const float* bq       = (const float*)d_in[2];
  const float* Wk       = (const float*)d_in[3];
  const float* bk       = (const float*)d_in[4];
  const float* Wv       = (const float*)d_in[5];
  const float* bv       = (const float*)d_in[6];
  const float* scalar   = (const float*)d_in[7];
  const float* ln_g     = (const float*)d_in[8];
  const float* ln_b     = (const float*)d_in[9];
  const float* eW1      = (const float*)d_in[10];
  const float* eb1      = (const float*)d_in[11];
  const float* eW2      = (const float*)d_in[12];
  const float* eb2      = (const float*)d_in[13];
  const float* e_scalar = (const float*)d_in[14];
  const float* eln_g    = (const float*)d_in[15];
  const float* eln_b    = (const float*)d_in[16];
  float* out = (float*)d_out;

  // workspace carve-up (256-byte aligned slabs)
  char* wp = (char*)d_ws;
  auto alloc = [&](size_t bytes) -> char* {
    char* p = wp;
    wp += (bytes + 255) & ~(size_t)255;
    return p;
  };
  const size_t NTOK = (size_t)B_ * S_;
  u16*   xb   = (u16*)  alloc(NTOK * D_ * 2);
  u16*   wqt  = (u16*)  alloc((size_t)H_ * DA_ * D_ * 2);
  u16*   wkt  = (u16*)  alloc((size_t)H_ * DA_ * D_ * 2);
  u16*   wvt  = (u16*)  alloc((size_t)H_ * DA_ * D_ * 2);
  u16*   w1t  = (u16*)  alloc((size_t)E_ * DFF_ * D_ * 2);
  u16*   w2t  = (u16*)  alloc((size_t)E_ * D_ * DFF_ * 2);
  u16*   qb   = (u16*)  alloc((size_t)B_ * H_ * S_ * DA_ * 2);
  u16*   kb   = (u16*)  alloc((size_t)B_ * H_ * S_ * DA_ * 2);
  u16*   vtb  = (u16*)  alloc((size_t)B_ * H_ * DA_ * S_ * 2);
  float* aout = (float*)alloc(NTOK * D_ * 4);
  float* x1f  = (float*)alloc(NTOK * D_ * 4);
  u16*   x1b  = (u16*)  alloc(NTOK * D_ * 2);
  u16*   hb   = (u16*)  alloc(NTOK * DFF_ * 2);
  float* yf   = (float*)alloc(NTOK * D_ * 4);

  // 1) bf16 conversions / weight transposes
  {
    int n = (int)(NTOK * D_);
    f32_to_bf16_kernel<<<(n + 255) / 256, 256, 0, stream>>>(x, xb, n);
    int nq = H_ * D_ * DA_;
    transpose_bf16_kernel<<<(nq + 255) / 256, 256, 0, stream>>>(Wq, wqt, D_, DA_, H_);
    transpose_bf16_kernel<<<(nq + 255) / 256, 256, 0, stream>>>(Wk, wkt, D_, DA_, H_);
    transpose_bf16_kernel<<<(nq + 255) / 256, 256, 0, stream>>>(Wv, wvt, D_, DA_, H_);
    int n1 = E_ * D_ * DFF_;
    transpose_bf16_kernel<<<(n1 + 255) / 256, 256, 0, stream>>>(eW1, w1t, D_, DFF_, E_);
    transpose_bf16_kernel<<<(n1 + 255) / 256, 256, 0, stream>>>(eW2, w2t, DFF_, D_, E_);
  }

  // 2) QKV projections (WMMA + async LDS staging)
  qkv_kernel<<<dim3((B_ * S_) / 256, H_, 3), 256, 0, stream>>>(
      xb, wqt, wkt, wvt, bq, bk, bv, qb, kb, vtb);

  // 3) causal flash attention (WMMA + online softmax)
  attn_kernel<<<B_ * H_ * (S_ / 128), 256, 0, stream>>>(qb, kb, vtb, aout);

  // 4) residual + LN 1
  ln1_kernel<<<(int)NTOK, 256, 0, stream>>>(x, aout, scalar, ln_g, ln_b, x1f, x1b);

  // 5) experts (WMMA GEMMs + async LDS staging)
  expert1_kernel<<<dim3(DFF_ / 64, 512 / 256, B_ * E_), 256, 0, stream>>>(
      x1b, w1t, eb1, hb);
  expert2_kernel<<<dim3(D_ / 64, 512 / 256, B_ * E_), 256, 0, stream>>>(
      hb, w2t, eb2, yf);

  // 6) residual + per-expert LN 2 -> output
  ln2_kernel<<<(int)NTOK, 256, 0, stream>>>(x1f, yf, e_scalar, eln_g, eln_b, out);
}
